// RegionProposalNetworkWrapper_70970039599786
// MI455X (gfx1250) — matrix operand order repaired and checked by reading, not compiled, not run on hardware
//
#include <hip/hip_runtime.h>
#include <hip/hip_bf16.h>
#include <stdint.h>

// ---------------- problem constants (match reference) ----------------
#define IMG_W      1216.0f
#define IMG_H      800.0f
#define NUM_LVL    5
#define A_TOT      242991
#define BATCH_N    4
#define M_CAND     4741          // 1000*4 + 741
#define SORT_N     8192          // pad for bitonic sort
#define MP_SCAN    5120          // M_CAND padded to 10 chunks of 512
#define POST_N     1000
#define NMS_TH     0.7f
#define MIN_SZ     1e-3f
#define XCLIP      4.135166556742356f   // log(1000/16)

__constant__ int c_loff[NUM_LVL]  = {0, 182400, 228000, 239400, 242250};
__constant__ int c_ln[NUM_LVL]    = {182400, 45600, 11400, 2850, 741};
__constant__ int c_lk[NUM_LVL]    = {1000, 1000, 1000, 1000, 741};
__constant__ int c_lbase[NUM_LVL] = {0, 1000, 2000, 3000, 4000};

typedef __attribute__((ext_vector_type(16))) _Float16 v16h;
typedef __attribute__((ext_vector_type(8)))  float    v8f;
typedef unsigned int u32x4 __attribute__((ext_vector_type(4)));
typedef int          i32x8 __attribute__((ext_vector_type(8)));
typedef int          i32x4 __attribute__((ext_vector_type(4)));

__device__ __forceinline__ unsigned fkey(float x) {
    unsigned u = __float_as_uint(x);
    return (u & 0x80000000u) ? ~u : (u | 0x80000000u);
}

// ---------------- K0: zero outputs / counters / keep flags ----------------
__global__ void rpn_init(float* out, int* ctr_gt, int* ctr_eq, unsigned* maxb, int* kf) {
    int i = blockIdx.x * blockDim.x + threadIdx.x;
    int st = gridDim.x * blockDim.x;
    for (int t = i; t < BATCH_N * (POST_N * 4 + POST_N); t += st) out[t] = 0.0f;
    for (int t = i; t < BATCH_N * NUM_LVL; t += st) { ctr_gt[t] = 0; ctr_eq[t] = 0; }
    for (int t = i; t < BATCH_N; t += st) maxb[t] = 0u;
    for (int t = i; t < BATCH_N * MP_SCAN; t += st) kf[t] = 0;
}

// ---------------- K1: per (batch,level) radix-select k-th threshold ----------------
__global__ __launch_bounds__(256)
void rpn_topk_thresh(const float* __restrict__ obj,
                     unsigned* thresh, int* gcnt, int* need) {
    int g = blockIdx.x;                 // 0 .. B*NL-1
    int b = g / NUM_LVL, l = g % NUM_LVL;
    int off = c_loff[l], n = c_ln[l], k = c_lk[l];
    const float* p = obj + (size_t)b * A_TOT + off;
    int tid = threadIdx.x, bd = blockDim.x;

    __shared__ unsigned hist[256];
    __shared__ unsigned sh_pref;
    __shared__ int sh_rem;
    __shared__ unsigned sh_g;
    if (tid == 0) { sh_pref = 0u; sh_rem = k; sh_g = 0u; }
    __syncthreads();

    for (int r = 0; r < 4; ++r) {
        for (int t = tid; t < 256; t += bd) hist[t] = 0u;
        __syncthreads();
        unsigned pref = sh_pref;
        int shift = 24 - 8 * r;
        for (int t = tid; t < n; t += bd) {
            unsigned key = fkey(p[t]);
            bool match = (r == 0) || ((key >> (shift + 8)) == (pref >> (shift + 8)));
            if (match) atomicAdd(&hist[(key >> shift) & 255u], 1u);
        }
        __syncthreads();
        if (tid == 0) {
            int rem = sh_rem; unsigned gg = sh_g; int bin = 255;
            for (; bin >= 0; --bin) {
                unsigned c = hist[bin];
                if ((int)c >= rem) break;
                rem -= (int)c; gg += c;
            }
            if (bin < 0) bin = 0;  // should not happen
            sh_pref = pref | ((unsigned)bin << shift);
            sh_rem = rem; sh_g = gg;
        }
        __syncthreads();
    }
    if (tid == 0) { thresh[g] = sh_pref; gcnt[g] = (int)sh_g; need[g] = sh_rem; }
}

// ---------------- K2: gather selected, sigmoid + decode + clip ----------------
#define GSUB 8
__global__ __launch_bounds__(256)
void rpn_gather(const float* __restrict__ obj,
                const float* __restrict__ deltas,
                const float* __restrict__ anchors,
                const unsigned* __restrict__ thresh,
                const int* __restrict__ gcnt, const int* __restrict__ need,
                int* ctr_gt, int* ctr_eq,
                float* cscore, float* cbox, int* clevel, unsigned* maxb) {
    int g = blockIdx.x / GSUB, sub = blockIdx.x % GSUB;
    int b = g / NUM_LVL, l = g % NUM_LVL;
    int off = c_loff[l], n = c_ln[l], base = c_lbase[l];
    unsigned T = thresh[g];
    int G = gcnt[g], NEQ = need[g];
    const float* p = obj + (size_t)b * A_TOT + off;
    int start = sub * blockDim.x + threadIdx.x;
    int stride = GSUB * blockDim.x;

    for (int i = start; i < n; i += stride) {
        float ov = p[i];
        unsigned key = fkey(ov);
        int slot = -1;
        if (key > T) {
            slot = base + atomicAdd(&ctr_gt[g], 1);
        } else if (key == T) {
            int q = atomicAdd(&ctr_eq[g], 1);
            if (q < NEQ) slot = base + G + q;
        }
        if (slot >= 0) {
            int ai = off + i;
            float sc = 1.0f / (1.0f + __expf(-ov));
            const float* an = anchors + (size_t)ai * 4;
            const float* dl = deltas + ((size_t)b * A_TOT + ai) * 4;
            float w  = an[2] - an[0], h = an[3] - an[1];
            float cx = an[0] + 0.5f * w, cy = an[1] + 0.5f * h;
            float dx = dl[0], dy = dl[1];
            float dw = fminf(dl[2], XCLIP), dh = fminf(dl[3], XCLIP);
            float pcx = dx * w + cx, pcy = dy * h + cy;
            float pw = __expf(dw) * w, ph = __expf(dh) * h;
            float x1 = fminf(fmaxf(pcx - 0.5f * pw, 0.0f), IMG_W);
            float y1 = fminf(fmaxf(pcy - 0.5f * ph, 0.0f), IMG_H);
            float x2 = fminf(fmaxf(pcx + 0.5f * pw, 0.0f), IMG_W);
            float y2 = fminf(fmaxf(pcy + 0.5f * ph, 0.0f), IMG_H);
            bool valid = (x2 - x1 >= MIN_SZ) && (y2 - y1 >= MIN_SZ) && (sc >= 0.0f);
            size_t ci = (size_t)b * M_CAND + slot;
            cscore[ci] = valid ? sc : -__builtin_inff();
            cbox[ci * 4 + 0] = x1; cbox[ci * 4 + 1] = y1;
            cbox[ci * 4 + 2] = x2; cbox[ci * 4 + 3] = y2;
            clevel[ci] = l;
            atomicMax(&maxb[b], __float_as_uint(x2));
            atomicMax(&maxb[b], __float_as_uint(y2));
        }
    }
}

// ---------------- K3: per-image bitonic sort (desc by score) in LDS ----------------
__global__ __launch_bounds__(1024)
void rpn_sort(const float* __restrict__ cscore, float* sscore, int* sidx) {
    int b = blockIdx.x;
    int tid = threadIdx.x, bd = blockDim.x;
    __shared__ float ss[SORT_N];   // 32 KB
    __shared__ int   si[SORT_N];   // 32 KB
    for (int t = tid; t < SORT_N; t += bd) {
        if (t < M_CAND) { ss[t] = cscore[(size_t)b * M_CAND + t]; si[t] = t; }
        else            { ss[t] = -__builtin_inff();              si[t] = -1; }
    }
    __syncthreads();
    for (unsigned k = 2; k <= SORT_N; k <<= 1) {
        for (unsigned j = k >> 1; j > 0; j >>= 1) {
            for (unsigned i = tid; i < SORT_N; i += bd) {
                unsigned ixj = i ^ j;
                if (ixj > i) {
                    float a = ss[i], c = ss[ixj];
                    bool descSeg = ((i & k) == 0);
                    bool doswap = descSeg ? (a < c) : (a > c);
                    if (doswap) {
                        ss[i] = c; ss[ixj] = a;
                        int t = si[i]; si[i] = si[ixj]; si[ixj] = t;
                    }
                }
            }
            __syncthreads();
        }
    }
    for (int t = tid; t < SORT_N; t += bd) {
        sscore[(size_t)b * SORT_N + t] = ss[t];
        sidx[(size_t)b * SORT_N + t] = si[t];
    }
}

// ---------------- K3b: materialize sorted, level-offset boxes ----------------
__global__ __launch_bounds__(256)
void rpn_make_sbox(const int* __restrict__ sidx, const float* __restrict__ cbox,
                   const int* __restrict__ clevel, const unsigned* __restrict__ maxb,
                   float* sbox) {
    int b = blockIdx.x;
    float step = __uint_as_float(maxb[b]) + 1.0f;
    const float4* cb4 = reinterpret_cast<const float4*>(cbox);
    float4* sb4 = reinterpret_cast<float4*>(sbox);
    for (int j = threadIdx.x; j < M_CAND; j += blockDim.x) {
        int ci = sidx[(size_t)b * SORT_N + j];
        float4 r;
        if (ci >= 0) {
            float off = (float)clevel[(size_t)b * M_CAND + ci] * step;
            float4 bb = cb4[(size_t)b * M_CAND + ci];
            r.x = bb.x + off; r.y = bb.y + off; r.z = bb.z + off; r.w = bb.w + off;
        } else { r.x = r.y = r.z = r.w = 0.0f; }
        sb4[(size_t)b * M_CAND + j] = r;
    }
}

// ---------------- K4: iterative greedy NMS, keep flags ----------------
__global__ __launch_bounds__(512)
void rpn_nms(const float* __restrict__ sscore, const float* __restrict__ sbox,
             int* kf) {
    int b = blockIdx.x;
    int tid = threadIdx.x, bd = blockDim.x;
    __shared__ unsigned s_sup[MP_SCAN / 32];   // suppression bitset
    __shared__ float s_box[4];
    __shared__ int sh_keep, sh_done, sh_cnt;
    for (int t = tid; t < MP_SCAN / 32; t += bd) s_sup[t] = 0u;
    if (tid == 0) { sh_done = 0; sh_cnt = 0; }
    __syncthreads();

    const float4* sb4 = reinterpret_cast<const float4*>(sbox) + (size_t)b * M_CAND;
    for (int i = 0; i < M_CAND; ++i) {
        if (tid == 0) {
            int kp = 0;
            if (!((s_sup[i >> 5] >> (i & 31)) & 1u)) {
                float s = sscore[(size_t)b * SORT_N + i];
                if (s > -3.0e38f) kp = 1;
            }
            if (kp) {
                kf[(size_t)b * MP_SCAN + i] = 1;
                float4 bb = sb4[i];
                s_box[0] = bb.x; s_box[1] = bb.y; s_box[2] = bb.z; s_box[3] = bb.w;
                if (++sh_cnt >= POST_N) sh_done = 1;
            }
            sh_keep = kp;
        }
        __syncthreads();
        if (sh_keep) {
            float ax1 = s_box[0], ay1 = s_box[1], ax2 = s_box[2], ay2 = s_box[3];
            float areaA = (ax2 - ax1) * (ay2 - ay1);
            for (int j = i + 1 + tid; j < M_CAND; j += bd) {
                if (!((s_sup[j >> 5] >> (j & 31)) & 1u)) {
                    float4 bb = sb4[j];
                    float areaB = (bb.z - bb.x) * (bb.w - bb.y);
                    float lx = fmaxf(ax1, bb.x), ly = fmaxf(ay1, bb.y);
                    float rx = fminf(ax2, bb.z), ry = fminf(ay2, bb.w);
                    float ww = fmaxf(rx - lx, 0.0f), hh = fmaxf(ry - ly, 0.0f);
                    float inter = ww * hh;
                    float iou = inter / (areaA + areaB - inter + 1e-9f);
                    if (iou > NMS_TH) atomicOr(&s_sup[j >> 5], 1u << (j & 31));
                }
            }
        }
        __syncthreads();
        if (sh_done) break;
    }
}

// ---------------- K5: TDM-stage flags to LDS, WMMA prefix-sum, scatter ----------------
__global__ __launch_bounds__(32)
void rpn_scan_scatter(const int* __restrict__ kf, const int* __restrict__ sidx,
                      const float* __restrict__ sscore, const float* __restrict__ cbox,
                      float* out) {
    int b = blockIdx.x;
    int lane = threadIdx.x;                      // 0..31, single wave32
    __shared__ int   s_kf[MP_SCAN];              // 20 KB: keep flags
    __shared__ int   s_ex[MP_SCAN];              // 20 KB: exclusive scan
    __shared__ float s_rp[256];                  // 16x16 row prefixes
    __shared__ float s_rowoff[16];

    // --- Tensor Data Mover: async copy MP_SCAN dwords of keep flags into LDS ---
    {
        unsigned lds_off = (unsigned)(size_t)(void*)&s_kf[0];  // addr[31:0] = LDS offset
        unsigned long long ga = (unsigned long long)(size_t)(kf + (size_t)b * MP_SCAN);
        u32x4 g0; i32x8 g1; i32x4 g2 = {0, 0, 0, 0}; i32x4 g3 = {0, 0, 0, 0};
        g0.x = 1u;                                           // count=1 (valid D#)
        g0.y = lds_off;                                      // lds_addr
        g0.z = (unsigned)(ga & 0xFFFFFFFFull);               // global_addr[31:0]
        g0.w = (unsigned)((ga >> 32) & 0x01FFFFFFull) | (2u << 30);  // addr[56:32] | type=2
        g1[0] = (int)(2u << 16);                             // data_size=4B
        g1[1] = (int)((MP_SCAN & 0xFFFF) << 16);             // tensor_dim0[15:0]
        g1[2] = (int)((MP_SCAN >> 16) | (1u << 16));         // td0[31:16] | tensor_dim1=1
        g1[3] = (int)((MP_SCAN & 0xFFFF) << 16);             // td1[31:16]=0 | tile_dim0
        g1[4] = 1;                                           // tile_dim1=1, tile_dim2=0
        g1[5] = MP_SCAN;                                     // tensor_dim0_stride lo
        g1[6] = 0; g1[7] = 0;
#if __has_include(<hip/amd_detail/amd_gfx1250_TDM.h>)
        i32x8 g4 = {0, 0, 0, 0, 0, 0, 0, 0};
        __builtin_amdgcn_tensor_load_to_lds(g0, g1, g2, g3, g4, 0);
#else
        __builtin_amdgcn_tensor_load_to_lds(g0, g1, g2, g3, 0);
#endif
        __builtin_amdgcn_s_wait_tensorcnt(0);
    }
    __syncthreads();

    // B matrix: b[k][n] = 1 iff k <= 2n+1  (prefix sample at odd K positions)
    v16h bm;
    {
        int n = lane & 15, kb = (lane < 16) ? 0 : 8;
        for (int j = 0; j < 16; ++j)
            bm[j] = (_Float16)((((j < 8) ? kb + j : kb + j + 8) <= 2 * n + 1) ? 1.0f : 0.0f);
    }
    float carry = 0.0f;                          // meaningful in lane 0 only
    int m = lane & 15, hf = lane >> 4;

    for (int ch = 0; ch < MP_SCAN / 512; ++ch) {
        // A matrix: a[m][k] = flag[ch*512 + m*32 + k] per 16-bit A 16x32 VGPR layout
        v16h av;
        for (int v = 0; v < 8; ++v)
            for (int p = 0; p < 2; ++p) {
                int k = (v < 4) ? (2 * v + p + 8 * hf) : (2 * (v - 4) + p + 16 + 8 * hf);
                av[2 * v + p] = (_Float16)(float)s_kf[ch * 512 + m * 32 + k];
            }
        v8f cz = {};
        v8f d = __builtin_amdgcn_wmma_f32_16x16x32_f16(false, av, false, bm,
                                                       (short)0, cz, false, false);
        // D layout: lane holds D[r + 8*(lane/16)][lane%16] in element r
        for (int r = 0; r < 8; ++r)
            s_rp[(r + 8 * hf) * 16 + (lane & 15)] = d[r];
        __syncthreads();
        if (lane == 0) {
            float acc = carry;
            for (int mm = 0; mm < 16; ++mm) { s_rowoff[mm] = acc; acc += s_rp[mm * 16 + 15]; }
            carry = acc;
        }
        __syncthreads();
        for (int i = lane; i < 512; i += 32) {
            int mm = i >> 5, kk = i & 31;
            float ex;
            if (kk == 0)       ex = s_rowoff[mm];
            else if (kk & 1) { int nn = (kk - 1) >> 1;     // incl(2n) = rp[n] - x[2n+1], kk==2n+1
                               ex = s_rowoff[mm] + s_rp[mm * 16 + nn]
                                    - (float)s_kf[ch * 512 + mm * 32 + kk]; }
            else             { int nn = (kk >> 1) - 1;     // incl(2n+1) = rp[n]
                               ex = s_rowoff[mm] + s_rp[mm * 16 + nn]; }
            s_ex[ch * 512 + i] = (int)(ex + 0.5f);
        }
        __syncthreads();
    }

    // Scatter kept candidates (in sorted order) to output positions
    const float4* cb4 = reinterpret_cast<const float4*>(cbox);
    float4* ob4 = reinterpret_cast<float4*>(out);
    for (int j = lane; j < M_CAND; j += 32) {
        if (s_kf[j]) {
            int pos = s_ex[j];
            if (pos < POST_N) {
                int ci = sidx[(size_t)b * SORT_N + j];
                float sc = sscore[(size_t)b * SORT_N + j];
                ob4[(size_t)b * POST_N + pos] = cb4[(size_t)b * M_CAND + ci];
                out[(size_t)BATCH_N * POST_N * 4 + (size_t)b * POST_N + pos] = sc;
            }
        }
    }
}

// ---------------- host launcher ----------------
extern "C" void kernel_launch(void* const* d_in, const int* in_sizes, int n_in,
                              void* d_out, int out_size, void* d_ws, size_t ws_size,
                              hipStream_t stream) {
    (void)in_sizes; (void)n_in; (void)out_size; (void)ws_size;
    const float* obj     = (const float*)d_in[0];   // [B, A]
    const float* deltas  = (const float*)d_in[1];   // [B, A, 4]
    const float* anchors = (const float*)d_in[2];   // [A, 4]
    float* out = (float*)d_out;                     // boxes [B,1000,4] ++ scores [B,1000]

    char* w = (char*)d_ws;
    size_t cur = 0;
    auto alloc = [&](size_t bytes) { size_t o = cur; cur = (cur + bytes + 255) & ~(size_t)255; return o; };
    int*      ctr_gt = (int*)     (w + alloc(sizeof(int)      * BATCH_N * NUM_LVL));
    int*      ctr_eq = (int*)     (w + alloc(sizeof(int)      * BATCH_N * NUM_LVL));
    unsigned* thr    = (unsigned*)(w + alloc(sizeof(unsigned) * BATCH_N * NUM_LVL));
    int*      gcnt   = (int*)     (w + alloc(sizeof(int)      * BATCH_N * NUM_LVL));
    int*      need   = (int*)     (w + alloc(sizeof(int)      * BATCH_N * NUM_LVL));
    unsigned* maxb   = (unsigned*)(w + alloc(sizeof(unsigned) * BATCH_N));
    float*    cscore = (float*)   (w + alloc(sizeof(float)    * BATCH_N * M_CAND));
    float*    cbox   = (float*)   (w + alloc(sizeof(float)    * BATCH_N * M_CAND * 4));
    int*      clevel = (int*)     (w + alloc(sizeof(int)      * BATCH_N * M_CAND));
    float*    sscore = (float*)   (w + alloc(sizeof(float)    * BATCH_N * SORT_N));
    int*      sidx   = (int*)     (w + alloc(sizeof(int)      * BATCH_N * SORT_N));
    float*    sbox   = (float*)   (w + alloc(sizeof(float)    * BATCH_N * M_CAND * 4));
    int*      kf     = (int*)     (w + alloc(sizeof(int)      * BATCH_N * MP_SCAN));

    rpn_init<<<80, 256, 0, stream>>>(out, ctr_gt, ctr_eq, maxb, kf);
    rpn_topk_thresh<<<BATCH_N * NUM_LVL, 256, 0, stream>>>(obj, thr, gcnt, need);
    rpn_gather<<<BATCH_N * NUM_LVL * GSUB, 256, 0, stream>>>(
        obj, deltas, anchors, thr, gcnt, need, ctr_gt, ctr_eq,
        cscore, cbox, clevel, maxb);
    rpn_sort<<<BATCH_N, 1024, 0, stream>>>(cscore, sscore, sidx);
    rpn_make_sbox<<<BATCH_N, 256, 0, stream>>>(sidx, cbox, clevel, maxb, sbox);
    rpn_nms<<<BATCH_N, 512, 0, stream>>>(sscore, sbox, kf);
    rpn_scan_scatter<<<BATCH_N, 32, 0, stream>>>(kf, sidx, sscore, cbox, out);
}
